// DiffusionTransformer_59562606461616
// MI455X (gfx1250) — matrix operand order repaired and checked by reading, not compiled, not run on hardware
//
// MI455X / gfx1250 implementation of the 8-layer Pairformer diffusion block.
// All GEMMs use v_wmma_f32_16x16x32_bf16 (bf16 in, f32 accumulate) with a
// register-staged double-buffered LDS pipeline.
#include <hip/hip_runtime.h>

typedef __attribute__((ext_vector_type(16))) __bf16 v16bf;
typedef __attribute__((ext_vector_type(8)))  float  v8f;
typedef __attribute__((ext_vector_type(4)))  __bf16 bf4;
typedef __attribute__((ext_vector_type(2)))  __bf16 bf2;

#define TOK 512
#define CH  768
#define NH  16
#define HDIM 48
#define CZD 32
#define NL  8
#define HIDD 1536

__device__ __forceinline__ float sigm(float x){ return 1.0f/(1.0f + __expf(-x)); }

// ---------------- row LayerNorm (no affine) ----------------
__global__ __launch_bounds__(256) void ln_rows(const float* __restrict__ x,
                                               float* __restrict__ y, int C)
{
  __shared__ float red[2][8];
  const int row = blockIdx.x;
  const float* xr = x + (size_t)row * C;
  float s = 0.f, ss = 0.f;
  for (int c = threadIdx.x; c < C; c += 256){ float v = xr[c]; s += v; ss += v*v; }
  #pragma unroll
  for (int off = 16; off; off >>= 1){ s += __shfl_xor(s, off, 32); ss += __shfl_xor(ss, off, 32); }
  if ((threadIdx.x & 31) == 0){ red[0][threadIdx.x >> 5] = s; red[1][threadIdx.x >> 5] = ss; }
  __syncthreads();
  float S = 0.f, SS = 0.f;
  #pragma unroll
  for (int i = 0; i < 8; i++){ S += red[0][i]; SS += red[1][i]; }
  const float mu  = S / C;
  const float var = SS / C - mu * mu;
  const float r   = rsqrtf(var + 1e-5f);
  float* yr = y + (size_t)row * C;
  for (int c = threadIdx.x; c < C; c += 256) yr[c] = (xr[c] - mu) * r;
}

// ---------------- WMMA GEMM: out = epilogue(A * W1^T [, A * W2^T]) ----------------
// A: [M][K] f32 (row major), W: [Nout][K] f32 (row major).  128x128x32 tiles,
// 8 waves (4x2), each wave = 2x4 tiles of 16x16 -> v_wmma_f32_16x16x32_bf16.
// Double-buffered LDS, register-staged global loads overlap with WMMA.
enum { M_PLAIN = 0, M_SIG = 1, M_ADALN = 2, M_SWIGLU = 3, M_GRES = 4 };

template<int MODE, int DUAL, int AMODE>
__global__ __launch_bounds__(256) void gemm_wmma(
    const float* __restrict__ A, const float* __restrict__ A2,
    const float* __restrict__ W1, const float* __restrict__ W2,
    const float* __restrict__ bias1, const float* __restrict__ aux,
    const float* __restrict__ res, float* __restrict__ out,
    int M, int Nout, int K)
{
  __shared__ __bf16 sA[2][128][40];        // [m][k], row padded -> conflict-free b128 reads
  __shared__ __bf16 sB[2][2][32][136];     // [k][n], row padded; read one bf16 element per lane

  const int t    = threadIdx.x;
  const int m0   = blockIdx.y * 128;
  const int n0   = blockIdx.x * 128;
  const int lane = t & 31;
  const int wid  = t >> 5;
  const int wm   = wid >> 1;               // 0..3 : 32-row strip
  const int wn   = wid & 1;                // 0..1 : 64-col strip

  // staging thread mapping (A: 4 quads; B: 2 slots of a 2-column pair x 4 k)
  const int arow = (t + 0 * 256) >> 3;           // base row for i=0; rows step by 32
  const int ac4  = (t & 7) << 2;
  const int bnp  = (t >> 3) << 1;                // n pair base (0..62 step 2); +64 for slot 1
  const int bkq  = (t & 7) << 2;                 // k quad

  const v8f vzero = {0.f,0.f,0.f,0.f,0.f,0.f,0.f,0.f};
  v8f acc [2][4];
  v8f accB[2][4];
  #pragma unroll
  for (int i = 0; i < 2; i++)
    #pragma unroll
    for (int j = 0; j < 4; j++){ acc[i][j] = vzero; accB[i][j] = vzero; }

  union F { struct { uint4 lo, hi; } u; v16bf v; };

  float4 ra[4];                    // A stage: 4 quads
  float4 rb[1 + DUAL][2][2];       // B stage: [mat][slot][row of pair]

  // ---- global -> registers for K-tile at k0 (with A-side fusion)
  auto load_tile = [&](int k0){
    #pragma unroll
    for (int i = 0; i < 4; i++){
      const int row = arow + i * 32;
      float4 f = *(const float4*)(A + (size_t)(m0 + row) * K + k0 + ac4);
      if constexpr (AMODE == 1){                       // elementwise A * A2 (g*o)
        float4 g4 = *(const float4*)(A2 + (size_t)(m0 + row) * K + k0 + ac4);
        f.x *= g4.x; f.y *= g4.y; f.z *= g4.z; f.w *= g4.w;
      } else if constexpr (AMODE == 2){                // row-scale by K-vector (s_norm_w)
        float4 s4 = *(const float4*)(A2 + k0 + ac4);
        f.x *= s4.x; f.y *= s4.y; f.z *= s4.z; f.w *= s4.w;
      }
      ra[i] = f;
    }
    #pragma unroll
    for (int bm = 0; bm < 1 + DUAL; bm++){
      const float* Wp = bm ? W2 : W1;
      #pragma unroll
      for (int sl = 0; sl < 2; sl++){
        const int n = bnp + sl * 64;
        rb[bm][sl][0] = *(const float4*)(Wp + (size_t)(n0 + n + 0) * K + k0 + bkq);
        rb[bm][sl][1] = *(const float4*)(Wp + (size_t)(n0 + n + 1) * K + k0 + bkq);
      }
    }
  };

  // ---- registers -> LDS (f32 -> bf16; B transposed to [k][n] with packed b32 stores)
  auto store_tile = [&](int buf){
    #pragma unroll
    for (int i = 0; i < 4; i++){
      const int row = arow + i * 32;
      bf4 h; h.x = (__bf16)ra[i].x; h.y = (__bf16)ra[i].y;
             h.z = (__bf16)ra[i].z; h.w = (__bf16)ra[i].w;
      *(bf4*)&sA[buf][row][ac4] = h;
    }
    #pragma unroll
    for (int bm = 0; bm < 1 + DUAL; bm++){
      #pragma unroll
      for (int sl = 0; sl < 2; sl++){
        const int n = bnp + sl * 64;
        const float* f0 = (const float*)&rb[bm][sl][0];
        const float* f1 = (const float*)&rb[bm][sl][1];
        #pragma unroll
        for (int j = 0; j < 4; j++){
          bf2 p; p.x = (__bf16)f0[j]; p.y = (__bf16)f1[j];
          *(bf2*)&sB[buf][bm][bkq + j][n] = p;       // two adjacent n, one ds_store_b32
        }
      }
    }
  };

  // ---- compute one K-tile from LDS buffer `cur`
  auto compute_tile = [&](int cur){
    // A fragments (16x32 bf16): lane<16 -> K 0-7 & 16-23 ; lane>=16 -> K 8-15 & 24-31
    v16bf af[2];
    #pragma unroll
    for (int mt = 0; mt < 2; mt++){
      const int mrow = wm * 32 + mt * 16 + (lane & 15);
      const int kb   = (lane >> 4) << 3;
      F f;
      f.u.lo = *(const uint4*)&sA[cur][mrow][kb];
      f.u.hi = *(const uint4*)&sA[cur][mrow][kb + 16];
      af[mt] = f.v;
    }
    // batch-load all B fragments, then WMMA burst (amortized s_wait_dscnt)
    v16bf bfrag[4];
    #pragma unroll
    for (int nt = 0; nt < 4; nt++){
      const int cb = wn * 64 + nt * 16;
      const int kb = (lane >> 4) << 3;
      #pragma unroll
      for (int s = 0; s < 16; s++)
        bfrag[nt][s] = sB[cur][0][kb + (s & 7) + ((s >> 3) << 4)][cb + (lane & 15)];
    }
    #pragma unroll
    for (int nt = 0; nt < 4; nt++)
      #pragma unroll
      for (int mt = 0; mt < 2; mt++)
        acc[mt][nt] = __builtin_amdgcn_wmma_f32_16x16x32_bf16(
            false, af[mt], false, bfrag[nt], (short)0, acc[mt][nt], false, false);
    if constexpr (DUAL){
      #pragma unroll
      for (int nt = 0; nt < 4; nt++){
        const int cb = wn * 64 + nt * 16;
        const int kb = (lane >> 4) << 3;
        #pragma unroll
        for (int s = 0; s < 16; s++)
          bfrag[nt][s] = sB[cur][1][kb + (s & 7) + ((s >> 3) << 4)][cb + (lane & 15)];
      }
      #pragma unroll
      for (int nt = 0; nt < 4; nt++)
        #pragma unroll
        for (int mt = 0; mt < 2; mt++)
          accB[mt][nt] = __builtin_amdgcn_wmma_f32_16x16x32_bf16(
              false, af[mt], false, bfrag[nt], (short)0, accB[mt][nt], false, false);
    }
  };

  // ---- software pipeline: prologue + (load k+1 | compute k | store k+1) loop
  const int KT = K >> 5;
  load_tile(0);
  store_tile(0);
  __syncthreads();
  for (int kt = 0; kt < KT; ++kt){
    const int cur = kt & 1;
    const bool more = (kt + 1) < KT;
    if (more) load_tile((kt + 1) << 5);      // global loads overlap the WMMA burst
    compute_tile(cur);
    if (more) store_tile(cur ^ 1);
    __syncthreads();
  }

  // ---- epilogue: C/D layout -> lane&15 = N, vgpr r = M (+8 for lanes 16-31)
  #pragma unroll
  for (int mt = 0; mt < 2; mt++){
    #pragma unroll
    for (int nt = 0; nt < 4; nt++){
      const int ng  = n0 + wn * 64 + nt * 16 + (lane & 15);
      const int mg0 = m0 + wm * 32 + mt * 16 + ((lane >> 4) << 3);
      #pragma unroll
      for (int r = 0; r < 8; r++){
        const int    mg = mg0 + r;
        const size_t oi = (size_t)mg * Nout + ng;
        const float  x1 = acc[mt][nt][r];
        float v;
        if constexpr (MODE == M_PLAIN){
          v = x1 + (bias1 ? bias1[ng] : 0.f);
        } else if constexpr (MODE == M_SIG){
          v = sigm(x1 + (bias1 ? bias1[ng] : 0.f));
        } else if constexpr (MODE == M_ADALN){
          v = sigm(x1 + bias1[ng]) * aux[oi] + accB[mt][nt][r];
        } else if constexpr (MODE == M_SWIGLU){
          v = x1 * sigm(x1) * accB[mt][nt][r];
        } else {  // M_GRES: out = res + aux*acc   (safe in-place: 1 thread per element)
          v = res[oi] + aux[oi] * x1;
        }
        out[oi] = v;
      }
    }
  }
}

// ---------------- pair-bias: pb[h][i][j] = LN(z_l[i,j,:]) @ wz[h,:] ----------------
__global__ __launch_bounds__(256) void pb_kernel(
    const float* __restrict__ zb,   // bias + l*CZ
    const float* __restrict__ znw, const float* __restrict__ znb,
    const float* __restrict__ wz, float* __restrict__ pb)
{
  const int tid = blockIdx.x * 256 + threadIdx.x;      // i*512 + j
  const float* zp = zb + (size_t)tid * (NL * CZD);
  float z[CZD];
  #pragma unroll
  for (int i = 0; i < 8; i++){
    float4 f = *(const float4*)(zp + i * 4);
    z[i*4+0] = f.x; z[i*4+1] = f.y; z[i*4+2] = f.z; z[i*4+3] = f.w;
  }
  float s = 0.f, ss = 0.f;
  #pragma unroll
  for (int c = 0; c < CZD; c++){ s += z[c]; ss += z[c]*z[c]; }
  const float mu  = s * (1.f/CZD);
  const float var = ss * (1.f/CZD) - mu*mu;
  const float r   = rsqrtf(var + 1e-5f);
  float y[CZD];
  #pragma unroll
  for (int c = 0; c < CZD; c++) y[c] = (z[c]-mu)*r*znw[c] + znb[c];
  #pragma unroll
  for (int h = 0; h < NH; h++){
    float a = 0.f;
    #pragma unroll
    for (int c = 0; c < CZD; c++) a += y[c] * wz[h*CZD + c];
    pb[(size_t)h * (TOK*TOK) + tid] = a;
  }
}

// ---------------- attention: per (head, 64-row block); K^T,V in LDS as bf16 ----------------
#define KT_STRIDE 520
__global__ __launch_bounds__(256) void attn_kernel(
    const float* __restrict__ q, const float* __restrict__ k,
    const float* __restrict__ v, const float* __restrict__ pb,
    const unsigned char* __restrict__ mask, float* __restrict__ o)
{
  extern __shared__ char smem[];
  __bf16* KT = (__bf16*)smem;                      // [48][520]  (k transposed, padded)
  __bf16* VS = KT + HDIM * KT_STRIDE;              // [512][48]
  float*  P  = (float*)(VS + (size_t)TOK * HDIM);  // [8 waves][512]

  const int h  = blockIdx.x;
  const int ib = blockIdx.y;
  for (int j = threadIdx.x; j < TOK; j += 256){
    const float* kr = k + (size_t)j * CH + h * HDIM;
    const float* vr = v + (size_t)j * CH + h * HDIM;
    #pragma unroll
    for (int c = 0; c < HDIM; c++){
      KT[c * KT_STRIDE + j] = (__bf16)kr[c];
      VS[j * HDIM + c]      = (__bf16)vr[c];
    }
  }
  __syncthreads();

  const int wid  = threadIdx.x >> 5;
  const int lane = threadIdx.x & 31;
  float* Pw = P + wid * TOK;
  const float scale = 0.14433756729740643f;        // 1/sqrt(48)
  const float* pbh  = pb + (size_t)h * TOK * TOK;

  for (int rr = 0; rr < 8; rr++){
    const int i = ib * 64 + wid * 8 + rr;
    float qr[HDIM];
    {
      const float* qp = q + (size_t)i * CH + h * HDIM;
      #pragma unroll
      for (int c = 0; c < HDIM; c++) qr[c] = qp[c];
    }
    float lg[16];
    float mx = -3.0e38f;
    #pragma unroll 4
    for (int jj = 0; jj < 16; jj++){
      const int j = jj * 32 + lane;
      float d = 0.f;
      #pragma unroll
      for (int c = 0; c < HDIM; c++) d += qr[c] * (float)KT[c * KT_STRIDE + j];
      d = d * scale + pbh[(size_t)i * TOK + j];
      if (!mask[j]) d -= 1e9f;
      lg[jj] = d;
      mx = fmaxf(mx, d);
    }
    #pragma unroll
    for (int off = 16; off; off >>= 1) mx = fmaxf(mx, __shfl_xor(mx, off, 32));
    float sum = 0.f;
    #pragma unroll
    for (int jj = 0; jj < 16; jj++){
      const float p = __expf(lg[jj] - mx);
      sum += p;
      Pw[jj * 32 + lane] = p;
    }
    #pragma unroll
    for (int off = 16; off; off >>= 1) sum += __shfl_xor(sum, off, 32);
    const float inv = 1.f / sum;
    #pragma unroll
    for (int pass = 0; pass < 2; pass++){
      const int c = pass * 32 + lane;
      if (c < HDIM){
        float accf = 0.f;
        #pragma unroll 8
        for (int j = 0; j < TOK; j++) accf += Pw[j] * (float)VS[j * HDIM + c];
        o[(size_t)i * CH + h * HDIM + c] = accf * inv;
      }
    }
  }
}

// ---------------- host orchestration ----------------
extern "C" void kernel_launch(void* const* d_in, const int* in_sizes, int n_in,
                              void* d_out, int out_size, void* d_ws, size_t ws_size,
                              hipStream_t stream)
{
  const float* a    = (const float*)d_in[0];
  const float* s    = (const float*)d_in[1];
  const float* bias = (const float*)d_in[2];
  const unsigned char* mask = (const unsigned char*)d_in[3];
  const float* ad_snw = (const float*)d_in[4];
  const float* ad_ssw = (const float*)d_in[5];
  const float* ad_ssb = (const float*)d_in[6];
  const float* ad_sbw = (const float*)d_in[7];
  const float* wq  = (const float*)d_in[8];
  const float* bq  = (const float*)d_in[9];
  const float* wk  = (const float*)d_in[10];
  const float* wv  = (const float*)d_in[11];
  const float* wg  = (const float*)d_in[12];
  const float* znw = (const float*)d_in[13];
  const float* znb = (const float*)d_in[14];
  const float* wz  = (const float*)d_in[15];
  const float* wo  = (const float*)d_in[16];
  const float* opw = (const float*)d_in[17];
  const float* opb = (const float*)d_in[18];
  const float* t_snw = (const float*)d_in[19];
  const float* t_ssw = (const float*)d_in[20];
  const float* t_ssb = (const float*)d_in[21];
  const float* t_sbw = (const float*)d_in[22];
  const float* gw   = (const float*)d_in[23];
  const float* a2b  = (const float*)d_in[24];
  const float* b2a  = (const float*)d_in[25];
  const float* topw = (const float*)d_in[26];
  const float* topb = (const float*)d_in[27];

  float* W = (float*)d_ws;
  const size_t A1 = (size_t)TOK * CH;
  float* ln_s  = W;
  float* aln   = W + 1 * A1;
  float* bwork = W + 2 * A1;
  float* q     = W + 3 * A1;
  float* kk    = W + 4 * A1;
  float* vv    = W + 5 * A1;
  float* gg    = W + 6 * A1;
  float* oo    = W + 7 * A1;
  float* gate  = W + 8 * A1;
  float* a_cur = W + 9 * A1;
  float* pb    = W + 10 * A1;                      // NH*TOK*TOK floats
  float* bb    = pb + (size_t)NH * TOK * TOK;      // TOK*HIDD floats

  const dim3 blk(256);
  const dim3 g768(CH / 128, TOK / 128);
  const dim3 g1536(HIDD / 128, TOK / 128);
  const size_t attn_lds = (size_t)(HDIM * KT_STRIDE + TOK * HDIM) * 2 + 8 * TOK * 4;

  ln_rows<<<TOK, blk, 0, stream>>>(s, ln_s, CH);
  hipMemcpyAsync(a_cur, a, A1 * sizeof(float), hipMemcpyDeviceToDevice, stream);

  for (int l = 0; l < NL; l++){
    const size_t W2o = (size_t)l * CH * CH;
    // ---- adaLN(a, s) -> bwork
    ln_rows<<<TOK, blk, 0, stream>>>(a_cur, aln, CH);
    gemm_wmma<M_ADALN, 1, 2><<<g768, blk, 0, stream>>>(
        ln_s, ad_snw + (size_t)l * CH, ad_ssw + W2o, ad_sbw + W2o,
        ad_ssb + (size_t)l * CH, aln, nullptr, bwork, TOK, CH, CH);
    // ---- q, k, v, gate projections
    gemm_wmma<M_PLAIN, 0, 0><<<g768, blk, 0, stream>>>(bwork, nullptr, wq + W2o, nullptr,
        bq + (size_t)l * CH, nullptr, nullptr, q, TOK, CH, CH);
    gemm_wmma<M_PLAIN, 0, 0><<<g768, blk, 0, stream>>>(bwork, nullptr, wk + W2o, nullptr,
        nullptr, nullptr, nullptr, kk, TOK, CH, CH);
    gemm_wmma<M_PLAIN, 0, 0><<<g768, blk, 0, stream>>>(bwork, nullptr, wv + W2o, nullptr,
        nullptr, nullptr, nullptr, vv, TOK, CH, CH);
    gemm_wmma<M_SIG, 0, 0><<<g768, blk, 0, stream>>>(bwork, nullptr, wg + W2o, nullptr,
        nullptr, nullptr, nullptr, gg, TOK, CH, CH);
    // ---- pair bias (single streaming pass over the 32 MiB bias slice)
    pb_kernel<<<TOK * TOK / 256, blk, 0, stream>>>(bias + (size_t)l * CZD,
        znw + (size_t)l * CZD, znb + (size_t)l * CZD, wz + (size_t)l * NH * CZD, pb);
    // ---- attention
    attn_kernel<<<dim3(NH, TOK / 64), blk, attn_lds, stream>>>(q, kk, vv, pb, mask, oo);
    // ---- gated output + residual:  a += sigmoid(s@opw^T+opb) * ((g*o)@wo^T)
    gemm_wmma<M_SIG, 0, 0><<<g768, blk, 0, stream>>>(s, nullptr, opw + W2o, nullptr,
        opb + (size_t)l * CH, nullptr, nullptr, gate, TOK, CH, CH);
    gemm_wmma<M_GRES, 0, 1><<<g768, blk, 0, stream>>>(oo, gg, wo + W2o, nullptr,
        nullptr, gate, a_cur, a_cur, TOK, CH, CH);
    // ---- transition: t = adaLN(a, s) ; bb = silu(t@gw^T) * (t@a2b^T)
    ln_rows<<<TOK, blk, 0, stream>>>(a_cur, aln, CH);
    gemm_wmma<M_ADALN, 1, 2><<<g768, blk, 0, stream>>>(
        ln_s, t_snw + (size_t)l * CH, t_ssw + W2o, t_sbw + W2o,
        t_ssb + (size_t)l * CH, aln, nullptr, bwork, TOK, CH, CH);
    gemm_wmma<M_SWIGLU, 1, 0><<<g1536, blk, 0, stream>>>(bwork, nullptr,
        gw + (size_t)l * HIDD * CH, a2b + (size_t)l * HIDD * CH,
        nullptr, nullptr, nullptr, bb, TOK, HIDD, CH);
    // ---- a += sigmoid(s@topw^T+topb) * (bb@b2a^T)
    gemm_wmma<M_SIG, 0, 0><<<g768, blk, 0, stream>>>(s, nullptr, topw + W2o, nullptr,
        topb + (size_t)l * CH, nullptr, nullptr, gate, TOK, CH, CH);
    gemm_wmma<M_GRES, 0, 0><<<g768, blk, 0, stream>>>(bb, nullptr,
        b2a + (size_t)l * CH * HIDD, nullptr, nullptr, gate, a_cur, a_cur, TOK, CH, HIDD);
  }
  hipMemcpyAsync(d_out, a_cur, A1 * sizeof(float), hipMemcpyDeviceToDevice, stream);
}
